// AttentionDecoder_44813688767117
// MI455X (gfx1250) — matrix operand order repaired and checked by reading, not compiled, not run on hardware
//
#include <hip/hip_runtime.h>
#include <stdint.h>

// Problem dims (fixed by the reference)
#define B  32
#define S  128
#define H  512
#define E  256
#define V  32000
#define T  64
#define XDIM (H + E)   // 768
#define G3   (3 * H)   // 1536
#define KT   (H / 32)  // 16 k-tiles of 32
#define NT   (V / 16)  // 2000 n-tiles of 16

typedef __attribute__((ext_vector_type(16))) __bf16 v16bf;
typedef __attribute__((ext_vector_type(8)))  float  v8f;

__device__ __forceinline__ uint32_t f2bf_rne(float f) {
  uint32_t u = __float_as_uint(f);
  return (u + 0x7FFFu + ((u >> 16) & 1u)) >> 16;  // round-to-nearest-even bf16
}

// ---------------------------------------------------------------------------
// One-time: convert out_W [V,H] fp32 -> bf16, pre-packed into the
// v_wmma_f32_16x16x32_bf16 B-fragment order used here (contiguous k, same as the A loads):
//   per (ntile, ktile): 32 lanes x 8 dwords; lane l covers column n=ntile*16+(l&15),
//   K range kb=(l>>4)*16, dword j = {bf16(K=kb+2j), bf16(K=kb+2j+1)}.
// ---------------------------------------------------------------------------
__global__ void k_pack_w(const float* __restrict__ outW, uint32_t* __restrict__ packed) {
  int id   = blockIdx.x * 256 + threadIdx.x;   // 0 .. NT*KT*32*8-1  (8,192,000)
  int j    = id & 7;
  int lane = (id >> 3) & 31;
  int tile = id >> 8;                          // ntile*KT + kt
  int kt   = tile % KT;
  int nt   = tile / KT;
  int n    = nt * 16 + (lane & 15);
  int k    = kt * 32 + ((lane >> 4) << 4) + 2 * j;
  const float* row = outW + (size_t)n * H;
  uint32_t lo = f2bf_rne(row[k]);
  uint32_t hi = f2bf_rne(row[k + 1]);
  packed[id] = lo | (hi << 16);
}

// h0 = 0, idx0 = SOS(1), sampled[:,0] = 1
__global__ void k_init_state(float* __restrict__ h, int* __restrict__ idx,
                             float* __restrict__ sampled) {
  int tid = threadIdx.x;
  for (int i = tid; i < B * H; i += 256) h[i] = 0.0f;
  if (tid < B) { idx[tid] = 1; sampled[(size_t)tid * T] = 1.0f; }
}

// dec_out[:,0,:] = onehot(1)
__global__ void k_init_sos(float* __restrict__ dec) {
  int id = blockIdx.x * 256 + threadIdx.x;     // B*V threads
  int b = id / V, v = id % V;
  dec[(size_t)b * T * V + v] = (v == 1) ? 1.0f : 0.0f;
}

// ---------------------------------------------------------------------------
// Step kernel 1: attention + context + embedding -> x[b, H+E]
// One block per batch row; operands LDS-resident.
// ---------------------------------------------------------------------------
__global__ void k_attn(const float* __restrict__ enc, const float* __restrict__ emb,
                       const float* __restrict__ attnW, const float* __restrict__ attnb,
                       const float* __restrict__ h, const int* __restrict__ idx,
                       float* __restrict__ x) {
  __shared__ float sh[H];
  __shared__ float sth[H];
  __shared__ float sc[S];
  __shared__ float smax, ssum;
  int b = blockIdx.x, tid = threadIdx.x;

  for (int i = tid; i < H; i += 256) sh[i] = h[(size_t)b * H + i];
  __syncthreads();
  // th = h @ attn_W.T + attn_b
  for (int j = tid; j < H; j += 256) {
    const float* wr = attnW + (size_t)j * H;
    float acc = attnb[j];
    for (int k = 0; k < H; ++k) acc += sh[k] * wr[k];
    sth[j] = acc;
  }
  __syncthreads();
  // scores[s] = enc[b,s,:] . th
  if (tid < S) {
    const float* er = enc + ((size_t)b * S + tid) * H;
    float acc = 0.0f;
    for (int k = 0; k < H; ++k) acc += er[k] * sth[k];
    sc[tid] = acc;
  }
  __syncthreads();
  if (tid == 0) {
    float m = sc[0];
    for (int s = 1; s < S; ++s) m = fmaxf(m, sc[s]);
    smax = m;
  }
  __syncthreads();
  if (tid < S) sc[tid] = __expf(sc[tid] - smax);
  __syncthreads();
  if (tid == 0) {
    float s = 0.0f;
    for (int i = 0; i < S; ++i) s += sc[i];
    ssum = s;
  }
  __syncthreads();
  float inv = 1.0f / ssum;
  // ctx = softmax(scores) @ enc[b]
  for (int hh = tid; hh < H; hh += 256) {
    const float* eb = enc + (size_t)b * S * H + hh;
    float acc = 0.0f;
    for (int s = 0; s < S; ++s) acc += sc[s] * eb[(size_t)s * H];
    x[(size_t)b * XDIM + hh] = acc * inv;
  }
  // x[:, H:] = emb[idx_c]
  int ix = idx[b];
  if (ix > V) ix = 3;
  const float* er = emb + (size_t)ix * E;
  for (int j = tid; j < E; j += 256) x[(size_t)b * XDIM + H + j] = er[j];
}

// ---------------------------------------------------------------------------
// Step kernel 2: GRU cell. h <- GRU(x, h); also emit bf16 copy of h for WMMA A.
// ---------------------------------------------------------------------------
__global__ void k_gru(const float* __restrict__ x, const float* __restrict__ wih,
                      const float* __restrict__ whh, const float* __restrict__ bih,
                      const float* __restrict__ bhh, float* __restrict__ h,
                      uint16_t* __restrict__ hbf) {
  __shared__ float sx[XDIM];
  __shared__ float sh[H];
  __shared__ float gi[G3];
  __shared__ float gh[G3];
  int b = blockIdx.x, tid = threadIdx.x;

  for (int i = tid; i < XDIM; i += 256) sx[i] = x[(size_t)b * XDIM + i];
  for (int i = tid; i < H; i += 256) sh[i] = h[(size_t)b * H + i];
  __syncthreads();
  for (int j = tid; j < G3; j += 256) {
    const float* wr = wih + (size_t)j * XDIM;
    float a = bih[j];
    for (int k = 0; k < XDIM; ++k) a += sx[k] * wr[k];
    gi[j] = a;
    const float* hr = whh + (size_t)j * H;
    float c = bhh[j];
    for (int k = 0; k < H; ++k) c += sh[k] * hr[k];
    gh[j] = c;
  }
  __syncthreads();
  for (int j = tid; j < H; j += 256) {
    float r  = 1.0f / (1.0f + __expf(-(gi[j] + gh[j])));
    float z  = 1.0f / (1.0f + __expf(-(gi[H + j] + gh[H + j])));
    float n  = tanhf(gi[2 * H + j] + r * gh[2 * H + j]);
    float hn = (1.0f - z) * n + z * sh[j];
    h[(size_t)b * H + j]  = hn;
    hbf[(size_t)b * H + j] = (uint16_t)f2bf_rne(hn);
  }
}

// ---------------------------------------------------------------------------
// Step kernel 3: logits[32, V] = h_bf16 @ out_W_bf16.T + out_b  via WMMA bf16.
// Grid: 250 blocks x 8 waves; each wave owns one 16-wide column tile and runs
// two accumulator chains (rows 0-15 and 16-31), sharing the B fragment.
// Writes raw logits into dec_out[:, t, :]; log-softmax fixes them up after.
// ---------------------------------------------------------------------------
__global__ void k_logits(const uint16_t* __restrict__ hbf,
                         const uint32_t* __restrict__ packedB,
                         const float* __restrict__ outb,
                         float* __restrict__ dec, int t) {
  int tid  = threadIdx.x;
  int wave = tid >> 5;
  int lane = tid & 31;
  int ntile = blockIdx.x * 8 + wave;           // 0..NT-1
  int ncol  = ntile * 16 + (lane & 15);

  v8f c0 = {}, c1 = {};
  int m    = lane & 15;
  int ksel = (lane >> 4) << 4;                 // A-fragment K base select (0 or 16)
  const uint16_t* arow0 = hbf + (size_t)m * H;         // batch rows 0-15
  const uint16_t* arow1 = hbf + (size_t)(m + 16) * H;  // batch rows 16-31
  const uint32_t* bbase = packedB + ((size_t)ntile * KT * 32 + lane) * 8;

  for (int kt = 0; kt < KT; ++kt) {
    union { v16bf v; uint4 q[2]; } a0, a1, bb;
    const uint4* bp = reinterpret_cast<const uint4*>(bbase + (size_t)kt * 256);
    bb.q[0] = bp[0];
    bb.q[1] = bp[1];
    int k0 = kt * 32 + ksel;
    a0.q[0] = *reinterpret_cast<const uint4*>(arow0 + k0);
    a0.q[1] = *reinterpret_cast<const uint4*>(arow0 + k0 + 8);
    a1.q[0] = *reinterpret_cast<const uint4*>(arow1 + k0);
    a1.q[1] = *reinterpret_cast<const uint4*>(arow1 + k0 + 8);
    // WGP-scope prefetch (locality 3 -> near scope) of the k-tile two steps
    // ahead: stages the L2-resident weight stream into the near cache with
    // real lead time over the pipelined loads above.
    if (kt + 2 < KT)
      __builtin_prefetch(bbase + (size_t)(kt + 2) * 256, 0, 3);
    c0 = __builtin_amdgcn_wmma_f32_16x16x32_bf16(false, a0.v, false, bb.v,
                                                 (short)0, c0, false, false);
    c1 = __builtin_amdgcn_wmma_f32_16x16x32_bf16(false, a1.v, false, bb.v,
                                                 (short)0, c1, false, false);
  }

  float bias = outb[ncol];
  int rbase = (lane >> 4) * 8;                 // C/D layout: VGPR i -> row i + 8*(lane>>4)
  for (int i = 0; i < 8; ++i) {
    int r = rbase + i;
    dec[((size_t)r        * T + t) * V + ncol] = c0[i] + bias;
    dec[((size_t)(r + 16) * T + t) * V + ncol] = c1[i] + bias;
  }
}

// ---------------------------------------------------------------------------
// Step kernel 4: in-place log-softmax of dec_out[b, t, :], argmax -> idx & sampled.
// One block per batch row.
// ---------------------------------------------------------------------------
__global__ void k_lsm(float* __restrict__ dec, float* __restrict__ sampled,
                      int* __restrict__ idx, int t) {
  __shared__ float sv[256];
  __shared__ int   si[256];
  int b = blockIdx.x, tid = threadIdx.x;
  float* row = dec + ((size_t)b * T + t) * V;

  float bm = -INFINITY; int bi = 0x7fffffff;
  for (int v = tid; v < V; v += 256) {
    float val = row[v];
    if (val > bm) { bm = val; bi = v; }       // first occurrence within thread
  }
  sv[tid] = bm; si[tid] = bi;
  __syncthreads();
  for (int s = 128; s > 0; s >>= 1) {
    if (tid < s) {
      if (sv[tid + s] > sv[tid] ||
          (sv[tid + s] == sv[tid] && si[tid + s] < si[tid])) {
        sv[tid] = sv[tid + s]; si[tid] = si[tid + s];
      }
    }
    __syncthreads();
  }
  float rmax = sv[0];
  int   amax = si[0];
  __syncthreads();

  float ls = 0.0f;
  for (int v = tid; v < V; v += 256) ls += __expf(row[v] - rmax);
  sv[tid] = ls;
  __syncthreads();
  for (int s = 128; s > 0; s >>= 1) {
    if (tid < s) sv[tid] += sv[tid + s];
    __syncthreads();
  }
  float lse = rmax + __logf(sv[0]);
  for (int v = tid; v < V; v += 256) row[v] -= lse;
  if (tid == 0) {
    idx[b] = amax;                             // greedy feedback for next step
    sampled[(size_t)b * T + t] = (float)amax;
  }
}

// ---------------------------------------------------------------------------
extern "C" void kernel_launch(void* const* d_in, const int* in_sizes, int n_in,
                              void* d_out, int out_size, void* d_ws, size_t ws_size,
                              hipStream_t stream) {
  (void)in_sizes; (void)n_in; (void)out_size; (void)ws_size;
  const float* enc   = (const float*)d_in[0];
  // d_in[1] inputs, d_in[2] final_encoder_hidden, d_in[3] max_length: unused by forward
  const float* emb   = (const float*)d_in[4];
  const float* attnW = (const float*)d_in[5];
  const float* attnb = (const float*)d_in[6];
  const float* wih   = (const float*)d_in[7];
  const float* whh   = (const float*)d_in[8];
  const float* bih   = (const float*)d_in[9];
  const float* bhh   = (const float*)d_in[10];
  const float* outW  = (const float*)d_in[11];
  const float* outb  = (const float*)d_in[12];

  float* dec     = (float*)d_out;                     // [B,T,V]
  float* sampled = dec + (size_t)B * T * V;           // [B,T,1] (written as floats)

  // Workspace layout (~33 MB total):
  char* ws = (char*)d_ws;
  uint32_t* packedB = (uint32_t*)ws;                                  // 32,768,000 B
  float*    h       = (float*)(ws + 32768000);                        //     65,536 B
  uint16_t* hbf     = (uint16_t*)(ws + 32768000 + 65536);             //     32,768 B
  int*      idx     = (int*)(ws + 32768000 + 65536 + 32768);          //        128 B
  float*    x       = (float*)(ws + 32768000 + 65536 + 32768 + 128);  //     98,304 B

  // One-time setup
  k_pack_w<<<NT * KT, 256, 0, stream>>>(outW, packedB);   // 32000 blocks
  k_init_state<<<1, 256, 0, stream>>>(h, idx, sampled);
  k_init_sos<<<(B * V) / 256, 256, 0, stream>>>(dec);     // 4000 blocks

  // Sequential greedy decode
  for (int t = 1; t < T; ++t) {
    k_attn  <<<B,      256, 0, stream>>>(enc, emb, attnW, attnb, h, idx, x);
    k_gru   <<<B,      256, 0, stream>>>(x, wih, whh, bih, bhh, h, hbf);
    k_logits<<<NT / 8, 256, 0, stream>>>(hbf, packedB, outb, dec, t);
    k_lsm   <<<B,      256, 0, stream>>>(dec, sampled, idx, t);
  }
}
